// DeepSeekMoE_82059645157465
// MI455X (gfx1250) — matrix operand order repaired and checked by reading, not compiled, not run on hardware
//
#include <hip/hip_runtime.h>
#include <math.h>

// ---------------------------------------------------------------------------
// DeepSeek-style MoE for MI455X (gfx1250), bf16 WMMA path.
// H=768, I=1536, E=7 routed (computed densely, weighted) + 1 shared, T=16384.
// ---------------------------------------------------------------------------

typedef __bf16 bf16_t;
typedef __attribute__((ext_vector_type(16))) __bf16 v16bf;
typedef __attribute__((ext_vector_type(8)))  float  v8f;

constexpr int kH = 768;
constexpr int kI = 1536;
constexpr int kE = 7;        // routed experts
constexpr int kT = 16384;    // B*S tokens
constexpr int kM = 32;       // tokens per workgroup (2 WMMA m-tiles)
constexpr int kLDX = kH + 8;   // padded bf16 row stride for x tile
constexpr int kLDH = kI + 8;   // padded bf16 row stride for hidden tile

// ---------------------------------------------------------------------------
// Router: logits = (x @ router_w) * routing_bias; sigmoid; top-2; normalize;
// scatter into cw[T][8]; cw[:,7] = 1.0 (shared-expert coefficient).
// ---------------------------------------------------------------------------
__global__ __launch_bounds__(256)
void router_kernel(const float* __restrict__ x,
                   const float* __restrict__ rw,     // [H, E] row-major
                   const float* __restrict__ rbias,  // [E]
                   float* __restrict__ cw)           // [T, 8]
{
    const int t = blockIdx.x * blockDim.x + threadIdx.x;
    if (t >= kT) return;

    float acc[kE];
#pragma unroll
    for (int e = 0; e < kE; ++e) acc[e] = 0.0f;

    const float* xr = x + (size_t)t * kH;
    for (int h = 0; h < kH; ++h) {
        const float xv = xr[h];
        const float* wr = rw + (size_t)h * kE;
#pragma unroll
        for (int e = 0; e < kE; ++e) acc[e] += xv * wr[e];
    }

    float p[kE];
#pragma unroll
    for (int e = 0; e < kE; ++e) {
        const float logit = acc[e] * rbias[e];   // NOTE: multiply, per reference
        p[e] = __builtin_amdgcn_rcpf(1.0f + __expf(-logit));
    }

    // top-2 (first-max-by-index on ties, matching jax top_k)
    int i0 = 0;
#pragma unroll
    for (int e = 1; e < kE; ++e) if (p[e] > p[i0]) i0 = e;
    int i1 = (i0 == 0) ? 1 : 0;
#pragma unroll
    for (int e = 0; e < kE; ++e) if (e != i0 && p[e] > p[i1]) i1 = e;

    const float inv = __builtin_amdgcn_rcpf(p[i0] + p[i1]);
    float* cwt = cw + (size_t)t * 8;
#pragma unroll
    for (int e = 0; e < kE; ++e) cwt[e] = 0.0f;
    cwt[i0] = p[i0] * inv;
    cwt[i1] = p[i1] * inv;
    cwt[7]  = 1.0f;   // shared expert weight
}

// ---------------------------------------------------------------------------
// WMMA fragment helpers (ISA cdna5 CDNA5 ISA layouts, wave32).
// A (16x32 bf16): lane (0..15,M=lane / 16..31,M=lane-16):
//   elems 0..7  -> K = k0 + half*8 + j
//   elems 8..15 -> K = k0 + 16 + half*8 + (j-8)
// B (32x16 bf16): lane: N = lane%16; elems j -> same K as A elem j
// C/D (16x16 f32): lane: N = lane%16; VGPR v -> M = v + 8*(lane/16)
// ---------------------------------------------------------------------------
__device__ __forceinline__ v8f wmma_bf16(v16bf a, v16bf b, v8f c) {
    return __builtin_amdgcn_wmma_f32_16x16x32_bf16(
        /*neg_a=*/false, a, /*neg_b=*/false, b,
        /*c_mod=*/(short)0, c, /*reuse_a=*/false, /*reuse_b=*/false);
}

__device__ __forceinline__ float fast_silu(float g) {
    // g * sigmoid(g), with single-instruction v_rcp_f32
    return g * __builtin_amdgcn_rcpf(1.0f + __expf(-g));
}

__device__ __forceinline__ v16bf load_a_lds(const bf16_t* s, int ld, int mbase,
                                            int k0, int lrow, int lhalf) {
    const bf16_t* p = s + (size_t)(mbase + lrow) * ld + k0 + lhalf * 8;
    v16bf a;
#pragma unroll
    for (int j = 0; j < 8; ++j) { a[j] = p[j]; a[j + 8] = p[16 + j]; }
    return a;
}

__device__ __forceinline__ v16bf load_b_f32(const float* __restrict__ W, int ldw,
                                            int k0, int n0, int lrow, int lhalf) {
    const float* p = W + (size_t)(k0 + lhalf * 8) * ldw + n0 + lrow;
    // Prefetch next 32-row K-block of this weight panel (global_prefetch_b8).
    __builtin_prefetch(p + (size_t)32 * ldw, 0, 3);
    v16bf b;
#pragma unroll
    for (int j = 0; j < 16; ++j) b[j] = (bf16_t)p[(size_t)(j < 8 ? j : j + 8) * ldw];
    return b;
}

// ---------------------------------------------------------------------------
// Fused MoE: per 32-token tile, loop 8 experts (7 routed + shared):
//   hidden = silu(x@Wg) * (x@Wu)  -> bf16 in LDS (never to HBM)
//   out   += coef[token] * (hidden @ Wd)  -> VGPR accumulators
// ---------------------------------------------------------------------------
__global__ __launch_bounds__(256, 2)
void moe_wmma_kernel(const float* __restrict__ x,
                     const float* __restrict__ cw,       // [T, 8]
                     const float* __restrict__ sw_gate,  // [H, I]
                     const float* __restrict__ sw_up,    // [H, I]
                     const float* __restrict__ sw_down,  // [I, H]
                     const float* __restrict__ rw_gate,  // [E, H, I]
                     const float* __restrict__ rw_up,    // [E, H, I]
                     const float* __restrict__ rw_down,  // [E, I, H]
                     float* __restrict__ out)            // [T, H]
{
    __shared__ bf16_t sx[kM * kLDX];     // x tile, bf16
    __shared__ bf16_t sh[kM * kLDH];     // hidden tile, bf16
    __shared__ float  scoef[kM];

    const int tid   = threadIdx.x;
    const int wid   = tid >> 5;          // wave 0..7
    const int lane  = tid & 31;
    const int lrow  = lane & 15;
    const int lhalf = lane >> 4;
    const int t0    = blockIdx.x * kM;

    // Stage 0: load + convert x tile to bf16 (coalesced, once per block).
    for (int i = tid; i < kM * kH; i += 256) {
        const int r = i / kH;
        const int c = i - r * kH;
        sx[r * kLDX + c] = (bf16_t)x[(size_t)(t0 + r) * kH + c];
    }

    const v8f vzero = {0.f, 0.f, 0.f, 0.f, 0.f, 0.f, 0.f, 0.f};
    v8f oacc[6][2];
#pragma unroll
    for (int nt = 0; nt < 6; ++nt) { oacc[nt][0] = vzero; oacc[nt][1] = vzero; }

    for (int ex = 0; ex < 8; ++ex) {
        const float* Wg; const float* Wu; const float* Wd;
        if (ex < kE) {
            Wg = rw_gate + (size_t)ex * kH * kI;
            Wu = rw_up   + (size_t)ex * kH * kI;
            Wd = rw_down + (size_t)ex * kI * kH;
        } else {
            Wg = sw_gate; Wu = sw_up; Wd = sw_down;
        }

        if (tid < kM) scoef[tid] = cw[(size_t)(t0 + tid) * 8 + ex];
        __syncthreads();   // x/coef ready; previous expert's hidden reads done

        // ---- Stage A: gate/up GEMMs (M=32, N=1536, K=768) + silu*up -> LDS
        for (int nt = 0; nt < 12; ++nt) {
            const int n0 = (wid * 12 + nt) * 16;
            v8f ag0 = vzero, ag1 = vzero, au0 = vzero, au1 = vzero;
#pragma unroll 1
            for (int k0 = 0; k0 < kH; k0 += 32) {
                const v16bf a0 = load_a_lds(sx, kLDX, 0,  k0, lrow, lhalf);
                const v16bf a1 = load_a_lds(sx, kLDX, 16, k0, lrow, lhalf);
                const v16bf bg = load_b_f32(Wg, kI, k0, n0, lrow, lhalf);
                const v16bf bu = load_b_f32(Wu, kI, k0, n0, lrow, lhalf);
                ag0 = wmma_bf16(a0, bg, ag0);
                ag1 = wmma_bf16(a1, bg, ag1);
                au0 = wmma_bf16(a0, bu, au0);
                au1 = wmma_bf16(a1, bu, au1);
            }
#pragma unroll
            for (int v = 0; v < 8; ++v) {
                const float h0 = fast_silu(ag0[v]) * au0[v];
                const float h1 = fast_silu(ag1[v]) * au1[v];
                sh[(v + 8 * lhalf) * kLDH      + n0 + lrow] = (bf16_t)h0;
                sh[(16 + v + 8 * lhalf) * kLDH + n0 + lrow] = (bf16_t)h1;
            }
        }
        __syncthreads();

        // ---- Stage B: down GEMM (M=32, N=768, K=1536), coef-scaled accumulate
        for (int nt = 0; nt < 6; ++nt) {
            const int n0 = (wid * 6 + nt) * 16;
            v8f d0 = vzero, d1 = vzero;
#pragma unroll 1
            for (int k0 = 0; k0 < kI; k0 += 32) {
                const v16bf a0 = load_a_lds(sh, kLDH, 0,  k0, lrow, lhalf);
                const v16bf a1 = load_a_lds(sh, kLDH, 16, k0, lrow, lhalf);
                const v16bf bd = load_b_f32(Wd, kH, k0, n0, lrow, lhalf);
                d0 = wmma_bf16(a0, bd, d0);
                d1 = wmma_bf16(a1, bd, d1);
            }
#pragma unroll
            for (int v = 0; v < 8; ++v) {
                oacc[nt][0][v] += scoef[v + 8 * lhalf]      * d0[v];
                oacc[nt][1][v] += scoef[16 + v + 8 * lhalf] * d1[v];
            }
        }
        __syncthreads();   // protect sh/scoef before next expert overwrites
    }

    // ---- Epilogue: write fp32 output tile
    for (int nt = 0; nt < 6; ++nt) {
        const int n0 = (wid * 6 + nt) * 16;
#pragma unroll
        for (int v = 0; v < 8; ++v) {
            out[(size_t)(t0 + v + 8 * lhalf) * kH + n0 + lrow]      = oacc[nt][0][v];
            out[(size_t)(t0 + 16 + v + 8 * lhalf) * kH + n0 + lrow] = oacc[nt][1][v];
        }
    }
}

// ---------------------------------------------------------------------------
extern "C" void kernel_launch(void* const* d_in, const int* in_sizes, int n_in,
                              void* d_out, int out_size, void* d_ws, size_t ws_size,
                              hipStream_t stream) {
    (void)in_sizes; (void)n_in; (void)out_size; (void)ws_size;

    const float* x            = (const float*)d_in[0];
    const float* router_w     = (const float*)d_in[1];
    const float* routing_bias = (const float*)d_in[2];
    const float* sw_gate      = (const float*)d_in[3];
    const float* sw_up        = (const float*)d_in[4];
    const float* sw_down      = (const float*)d_in[5];
    const float* rw_gate      = (const float*)d_in[6];
    const float* rw_up        = (const float*)d_in[7];
    const float* rw_down      = (const float*)d_in[8];
    float*       out          = (float*)d_out;
    float*       cw           = (float*)d_ws;   // [T, 8] floats = 512 KB

    router_kernel<<<kT / 256, 256, 0, stream>>>(x, router_w, routing_bias, cw);
    moe_wmma_kernel<<<kT / kM, 256, 0, stream>>>(x, cw, sw_gate, sw_up, sw_down,
                                                 rw_gate, rw_up, rw_down, out);
}
